// ViGBlock_47442208751884
// MI455X (gfx1250) — compile-verified, matrix operand order checked by reading, not run on hardware
//
#include <hip/hip_runtime.h>

// ---------------------------------------------------------------------------
// ViG block for MI455X (gfx1250): bf16 WMMA GEMMs for patch-embed conv and
// unpatch, async global->LDS copies for the weight tiles, fused LDS-resident
// minGRU log-space scan, grouped pos-conv add.
//
// Workspace layout (bytes, 256-aligned chunks), total ~508 MB:
//   xb   : x in bf16                  (8*96*256*256)*2   = 100,663,296
//   wrp  : r_pw bf16 flat (384x2400)  921600*2           =   1,843,200
//   wcp  : c_pw bf16 flat             921600*2           =   1,843,200
//   wur  : r_un_w bf16 flat           147456*2           =     294,912
//   wuc  : c_un_w bf16 flat           147456*2           =     294,912
//   rbuf : patch-conv r out bf16      (8*384*128*128)*2  = 100,663,296
//   cbuf : patch-conv c out bf16                          100,663,296
//   rh   : minGRU(r)+pe(r) bf16                           100,663,296
//   ch   : minGRU(c)+pe(c) bf16                           100,663,296
// ---------------------------------------------------------------------------

typedef __attribute__((ext_vector_type(16))) __bf16 bf16x16;
typedef __attribute__((ext_vector_type(8)))  float  f32x8;

#define LDSTR 48                 // LDS row stride in bf16 elems (96 B, 32B-aligned)

__device__ __forceinline__ unsigned short f2bf(float f) {
  __bf16 h = (__bf16)f;
  return __builtin_bit_cast(unsigned short, h);
}
__device__ __forceinline__ float bf2f(unsigned short u) {
  __bf16 h = __builtin_bit_cast(__bf16, u);
  return (float)h;
}
__device__ __forceinline__ float softplusf(float x) {
  return (x > 20.f) ? x : log1pf(__expf(x));
}
// LDS byte offset of a shared-memory pointer: the LDS aperture lives in the
// high 32 bits of the generic address, so the low 32 bits are the LDS offset.
__device__ __forceinline__ unsigned lds_off32(const void* p) {
  return (unsigned)(unsigned long long)(uintptr_t)p;
}

// ---------------------------------------------------------------------------
// fp32 -> bf16 conversion, float4 granularity (all sizes are multiples of 4)
// ---------------------------------------------------------------------------
__global__ __launch_bounds__(256)
void cvt_f32_bf16(const float* __restrict__ in, unsigned short* __restrict__ out,
                  int n4) {
  int i = blockIdx.x * 256 + threadIdx.x;
  if (i >= n4) return;
  float4 v = ((const float4*)in)[i];
  ushort4 o;
  o.x = f2bf(v.x); o.y = f2bf(v.y); o.z = f2bf(v.z); o.w = f2bf(v.w);
  ((ushort4*)out)[i] = o;
}

// ---------------------------------------------------------------------------
// Patch-embed conv as implicit-im2col WMMA GEMM.
//   out[m, (b,hp,wp)] = sum_k W[m,k] * im2col[k,(b,hp,wp)] + bias[m]
//   M=384 (grid.y tiles of 128), N tile = one (b,hp) row of 128 wp,
//   K = C*5*5 = 2400, stepped by 32 (v_wmma_f32_16x16x32_bf16).
// 8 waves per WG; wave (wm,wn) owns a 32x64 sub-tile -> 8 WMMAs / K-step.
// A tile (weights) goes global->LDS via the CDNA5 async pipe (ASYNCcnt);
// B tile is a strided im2col gather with incremental (c,kh,kw) decomposition.
// ---------------------------------------------------------------------------
__global__ __launch_bounds__(256)
void patch_gemm(const unsigned short* __restrict__ xb,     // (8,96,256,256) bf16
                const unsigned short* __restrict__ wflat,  // (384,2400) bf16
                const float* __restrict__ bias,            // (384)
                unsigned short* __restrict__ outbuf) {     // (8,384,128,128) bf16
  __shared__ __align__(64) unsigned short ldsA[128 * LDSTR];
  __shared__ __align__(64) unsigned short ldsB[128 * LDSTR];

  const int nt = blockIdx.x;          // (b,hp)
  const int b  = nt >> 7;
  const int hp = nt & 127;
  const int m_base = blockIdx.y * 128;

  const int tid  = threadIdx.x;
  const int lane = tid & 31;
  const int wave = tid >> 5;
  const int wm   = wave & 3;          // M sub-block (32 rows)
  const int wn   = wave >> 2;         // N sub-block (64 cols)

  // Per-thread im2col coordinates, advanced incrementally (k = kt*32 + kk).
  const int kk = tid >> 3;            // 0..31
  const int wg = tid & 7;             // wp group of 16
  int c   = (kk >= 25) ? 1 : 0;       // k / 25
  int rem = kk - 25 * c;              // k % 25

  // Async A-tile destination (fixed per thread) and weight row base.
  const unsigned ldsA_dst = lds_off32(&ldsA[(tid & 127) * LDSTR]);
  const unsigned short* wrow = wflat + (size_t)(m_base + (tid & 127)) * 2400;

  f32x8 acc[2][4] = {};

  for (int kt = 0; kt < 75; ++kt) {
    const int k0 = kt * 32;
    // ---- A tile: 128 M rows x 32 K, async global->LDS (64B per row) -------
    if (tid < 128) {
      unsigned long long gsrc = (unsigned long long)(uintptr_t)(wrow + k0);
      asm volatile(
          "global_load_async_to_lds_b128 %0, %1, off\n\t"
          "global_load_async_to_lds_b128 %0, %1, off offset:16\n\t"
          "global_load_async_to_lds_b128 %0, %1, off offset:32\n\t"
          "global_load_async_to_lds_b128 %0, %1, off offset:48"
          :: "v"(ldsA_dst), "v"(gsrc) : "memory");
      if (kt + 1 < 75)
        __builtin_prefetch(wrow + k0 + 32, 0, 3);     // global_prefetch_b8
    }
    // ---- B tile: implicit im2col gather, stored transposed [wp][k] -------
    {
      const int kh = (rem * 205) >> 10;               // rem / 5 (rem < 25)
      const int kw = rem - kh * 5;
      const int ih = 2 * hp + kh - 2;                 // stride 2, pad 2
      if (ih >= 0 && ih < 256) {
        const unsigned short* row = xb + (((size_t)b * 96 + c) * 256 + ih) * 256;
        #pragma unroll
        for (int i = 0; i < 16; ++i) {
          int wp = wg * 16 + i;
          int iw = 2 * wp + kw - 2;
          ldsB[wp * LDSTR + kk] = (iw >= 0 && iw < 256) ? row[iw] : (unsigned short)0;
        }
      } else {
        #pragma unroll
        for (int i = 0; i < 16; ++i)
          ldsB[(wg * 16 + i) * LDSTR + kk] = 0;
      }
      // advance k by 32: k%25 += 7 (mod 25), k/25 += 1 or 2
      rem += 7; ++c;
      if (rem >= 25) { rem -= 25; ++c; }
    }
    asm volatile("s_wait_asynccnt 0x0" ::: "memory"); // A-tile landed in LDS
    __syncthreads();
    // ---- fragments + 8 WMMAs ---------------------------------------------
    const int frow = lane & 15;
    const int koff = (lane >> 4) * 16;
    bf16x16 afrag[2], bfrag[4];
    #pragma unroll
    for (int i = 0; i < 2; ++i)
      afrag[i] = *(const bf16x16*)&ldsA[(wm * 32 + i * 16 + frow) * LDSTR + koff];
    #pragma unroll
    for (int j = 0; j < 4; ++j)
      bfrag[j] = *(const bf16x16*)&ldsB[(wn * 64 + j * 16 + frow) * LDSTR + koff];
    #pragma unroll
    for (int i = 0; i < 2; ++i)
      #pragma unroll
      for (int j = 0; j < 4; ++j)
        acc[i][j] = __builtin_amdgcn_wmma_f32_16x16x32_bf16(
            false, afrag[i], false, bfrag[j], (short)0, acc[i][j], false, false);
    __syncthreads();
  }

  // ---- epilogue: C/D layout VGPR v: M = v + 8*(lane>>4), N = lane&15 -----
  const int mrow = (lane >> 4) * 8;
  const int ncol = lane & 15;
  #pragma unroll
  for (int i = 0; i < 2; ++i)
    #pragma unroll
    for (int j = 0; j < 4; ++j) {
      const int wp = wn * 64 + j * 16 + ncol;
      #pragma unroll
      for (int v = 0; v < 8; ++v) {
        const int m = m_base + wm * 32 + i * 16 + mrow + v;
        float val = acc[i][j][v] + bias[m];
        outbuf[(((size_t)b * 384 + m) * 128 + hp) * 128 + wp] = f2bf(val);
      }
    }
}

// ---------------------------------------------------------------------------
// Fused minGRU: hg = seq @ W in-registers per step (W resident in LDS),
// streaming log-space heinsen scan (running cumsum + running logsumexp).
// One 96-thread block per (b, head, scan-line). Thread j owns feature j.
//   row scan: t = hp (s_t=128), fixed = wp (s_f=1)
//   col scan: t = wp (s_t=1),   fixed = hp (s_f=128)
// ---------------------------------------------------------------------------
__global__ __launch_bounds__(96)
void mingru_scan(const unsigned short* __restrict__ seqbuf,  // (8,384,128,128) bf16
                 const float* __restrict__ w,                // (96,192)
                 unsigned short* __restrict__ outbuf,        // same layout as seqbuf
                 int s_t, int s_f) {
  __shared__ float wlds[96 * 192];
  __shared__ float srow[96];
  const int j = threadIdx.x;                                 // feature 0..95
  for (int idx = j; idx < 96 * 192; idx += 96) wlds[idx] = w[idx];

  const int blk   = blockIdx.x;          // (b*4+head)*128 + fixed
  const int fixed = blk & 127;
  const int bh    = blk >> 7;            // b*4 + head
  const size_t base = (size_t)bh * 96 * 16384;
  __syncthreads();

  float a = 0.f, mrun = -1e30f, srun = 0.f;
  for (int t = 0; t < 128; ++t) {
    srow[j] = bf2f(seqbuf[base + (size_t)j * 16384 + (size_t)fixed * s_f + (size_t)t * s_t]);
    __syncthreads();
    float hid = 0.f, gat = 0.f;
    #pragma unroll 4
    for (int d = 0; d < 96; ++d) {
      float s = srow[d];
      hid = fmaf(s, wlds[d * 192 + j], hid);
      gat = fmaf(s, wlds[d * 192 + 96 + j], gat);
    }
    float log_coeff = -softplusf(gat);                       // log(1-z)
    float log_z     = -softplusf(-gat);                      // log z
    float log_th    = (hid >= 0.f) ? __logf(hid + 0.5f) : -softplusf(-hid);
    a += log_coeff;                                          // a_star (inclusive)
    float xv = log_z + log_th - a;
    float mn = fmaxf(mrun, xv);
    srun = srun * __expf(mrun - mn) + __expf(xv - mn);
    mrun = mn;
    float h = __expf(a + mrun + __logf(srun));
    outbuf[base + (size_t)j * 16384 + (size_t)fixed * s_f + (size_t)t * s_t] = f2bf(h);
    __syncthreads();
  }
}

// ---------------------------------------------------------------------------
// Grouped positional conv (groups=96, 4ch per group, 3x3, pad 1), added into
// the minGRU output buffer: dst += conv(src) + bias.
// ---------------------------------------------------------------------------
__global__ __launch_bounds__(256)
void pe_add(const unsigned short* __restrict__ src,   // (8,384,128,128) bf16
            const float* __restrict__ w,              // (384,4,3,3)
            const float* __restrict__ bias,           // (384)
            unsigned short* __restrict__ dst) {
  const size_t idx = (size_t)blockIdx.x * 256 + threadIdx.x;
  const int wp  = (int)(idx & 127);
  const int hp  = (int)((idx >> 7) & 127);
  const int tmp = (int)(idx >> 14);
  const int dch = tmp % 384;
  const int b   = tmp / 384;
  const int g   = dch >> 2;                            // group = d / (DIM/groups)
  float acc = bias[dch];
  const float* wk = w + dch * 36;
  const unsigned short* in0 = src + ((size_t)b * 384 + g * 4) * 16384;
  #pragma unroll
  for (int i = 0; i < 4; ++i)
    #pragma unroll
    for (int kh = 0; kh < 3; ++kh) {
      const int ih = hp + kh - 1;
      if (ih < 0 || ih >= 128) continue;
      #pragma unroll
      for (int kw = 0; kw < 3; ++kw) {
        const int iw = wp + kw - 1;
        if (iw < 0 || iw >= 128) continue;
        acc = fmaf(bf2f(in0[((size_t)i * 16384) + ih * 128 + iw]),
                   wk[i * 9 + kh * 3 + kw], acc);
      }
    }
  dst[idx] = f2bf(bf2f(dst[idx]) + acc);
}

// ---------------------------------------------------------------------------
// Unpatch (ConvTranspose2d k=2,s=2) for BOTH paths + final sum, as 4 WMMA
// GEMMs (one per (p,q)): out[o,(b,hp,wp)@pq] = sum_c Wpq[c,o]*term[c,(b,hp,wp)]
//   M=96 (6 tiles), N tile = one (b,hp) row of 128 wp (8 waves x 16 cols),
//   K=384 per path, both paths accumulated into the same f32 accumulators.
// Writes the final fp32 output exactly once per element.
// ---------------------------------------------------------------------------
__global__ __launch_bounds__(256)
void unpatch_out(const unsigned short* __restrict__ rterm,
                 const unsigned short* __restrict__ cterm,
                 const unsigned short* __restrict__ wr,   // (384,96,2,2) bf16
                 const unsigned short* __restrict__ wc,
                 const float* __restrict__ br,            // (96)
                 const float* __restrict__ bc,
                 float* __restrict__ out) {               // (8,96,256,256)
  __shared__ __align__(64) unsigned short ldsA[96 * LDSTR];
  __shared__ __align__(64) unsigned short ldsB[128 * LDSTR];
  const int nt = blockIdx.x;
  const int b  = nt >> 7;
  const int hp = nt & 127;
  const int tid = threadIdx.x, lane = tid & 31, wave = tid >> 5;

  #pragma unroll 1
  for (int p = 0; p < 2; ++p)
  #pragma unroll 1
  for (int q = 0; q < 2; ++q) {
    f32x8 acc[6] = {};
    #pragma unroll 1
    for (int path = 0; path < 2; ++path) {
      const unsigned short* term = path ? cterm : rterm;
      const unsigned short* wub  = path ? wc : wr;
      #pragma unroll 1
      for (int kt = 0; kt < 12; ++kt) {
        const int c0 = kt * 32;
        // A tile: [o][c]  from wub[(c*96+o)*4 + p*2 + q]
        if (tid < 96) {
          const int o = tid;
          #pragma unroll
          for (int kk = 0; kk < 32; ++kk)
            ldsA[o * LDSTR + kk] = wub[(size_t)(c0 + kk) * 384 + o * 4 + p * 2 + q];
        }
        // B tile: [wp][c]
        {
          const int kk = tid >> 3, wg = tid & 7;
          const unsigned short* srcrow =
              term + (((size_t)b * 384 + (c0 + kk)) * 128 + hp) * 128;
          #pragma unroll
          for (int i = 0; i < 16; ++i) {
            const int wp = wg * 16 + i;
            ldsB[wp * LDSTR + kk] = srcrow[wp];
          }
        }
        __syncthreads();
        const int frow = lane & 15;
        const int koff = (lane >> 4) * 16;
        bf16x16 bfrag = *(const bf16x16*)&ldsB[(wave * 16 + frow) * LDSTR + koff];
        #pragma unroll
        for (int mt = 0; mt < 6; ++mt) {
          bf16x16 afrag = *(const bf16x16*)&ldsA[(mt * 16 + frow) * LDSTR + koff];
          acc[mt] = __builtin_amdgcn_wmma_f32_16x16x32_bf16(
              false, afrag, false, bfrag, (short)0, acc[mt], false, false);
        }
        __syncthreads();
      }
    }
    // store: oh = 2*hp+p, ow = 2*wp+q
    const int mrow = (lane >> 4) * 8;
    const int wp   = wave * 16 + (lane & 15);
    const int oh   = 2 * hp + p;
    const int ow   = 2 * wp + q;
    #pragma unroll
    for (int mt = 0; mt < 6; ++mt)
      #pragma unroll
      for (int v = 0; v < 8; ++v) {
        const int o = mt * 16 + mrow + v;
        out[(((size_t)b * 96 + o) * 256 + oh) * 256 + ow] = acc[mt][v] + br[o] + bc[o];
      }
  }
}

// ---------------------------------------------------------------------------
extern "C" void kernel_launch(void* const* d_in, const int* in_sizes, int n_in,
                              void* d_out, int out_size, void* d_ws, size_t ws_size,
                              hipStream_t stream) {
  const float* x      = (const float*)d_in[0];
  const float* r_pw   = (const float*)d_in[1];
  const float* r_pb   = (const float*)d_in[2];
  const float* c_pw   = (const float*)d_in[3];
  const float* c_pb   = (const float*)d_in[4];
  const float* r_pe_w = (const float*)d_in[5];
  const float* r_pe_b = (const float*)d_in[6];
  const float* c_pe_w = (const float*)d_in[7];
  const float* c_pe_b = (const float*)d_in[8];
  const float* r_un_w = (const float*)d_in[9];
  const float* r_un_b = (const float*)d_in[10];
  const float* c_un_w = (const float*)d_in[11];
  const float* c_un_b = (const float*)d_in[12];
  const float* r_gruw = (const float*)d_in[13];
  const float* c_gruw = (const float*)d_in[14];
  float* out = (float*)d_out;

  const size_t NX  = (size_t)8 * 96 * 256 * 256;    // 50,331,648
  const size_t NPW = (size_t)384 * 96 * 25;         // 921,600
  const size_t NUN = (size_t)384 * 96 * 4;          // 147,456
  const size_t NR  = (size_t)8 * 384 * 128 * 128;   // 50,331,648

  unsigned char* ws = (unsigned char*)d_ws;
  size_t off = 0;
  auto alloc = [&](size_t bytes) -> void* {
    void* p = ws + off;
    off += (bytes + 255) & ~(size_t)255;
    return p;
  };
  unsigned short* xb   = (unsigned short*)alloc(NX * 2);
  unsigned short* wrp  = (unsigned short*)alloc(NPW * 2);
  unsigned short* wcp  = (unsigned short*)alloc(NPW * 2);
  unsigned short* wur  = (unsigned short*)alloc(NUN * 2);
  unsigned short* wuc  = (unsigned short*)alloc(NUN * 2);
  unsigned short* rbuf = (unsigned short*)alloc(NR * 2);
  unsigned short* cbuf = (unsigned short*)alloc(NR * 2);
  unsigned short* rh   = (unsigned short*)alloc(NR * 2);
  unsigned short* ch   = (unsigned short*)alloc(NR * 2);
  (void)ws_size; (void)n_in; (void)in_sizes; (void)out_size;

  // 1) fp32 -> bf16 conversions
  cvt_f32_bf16<<<(int)((NX / 4 + 255) / 256), 256, 0, stream>>>(x, xb, (int)(NX / 4));
  cvt_f32_bf16<<<(int)((NPW / 4 + 255) / 256), 256, 0, stream>>>(r_pw, wrp, (int)(NPW / 4));
  cvt_f32_bf16<<<(int)((NPW / 4 + 255) / 256), 256, 0, stream>>>(c_pw, wcp, (int)(NPW / 4));
  cvt_f32_bf16<<<(int)((NUN / 4 + 255) / 256), 256, 0, stream>>>(r_un_w, wur, (int)(NUN / 4));
  cvt_f32_bf16<<<(int)((NUN / 4 + 255) / 256), 256, 0, stream>>>(c_un_w, wuc, (int)(NUN / 4));

  // 2) patch-embed convs as WMMA GEMMs (M tiles: 384/128 = 3)
  dim3 pgrid(1024, 3);
  patch_gemm<<<pgrid, 256, 0, stream>>>(xb, wrp, r_pb, rbuf);
  patch_gemm<<<pgrid, 256, 0, stream>>>(xb, wcp, c_pb, cbuf);

  // 3) minGRU scans: row (over Hp) and col (over Wp)
  mingru_scan<<<4096, 96, 0, stream>>>(rbuf, r_gruw, rh, /*s_t=*/128, /*s_f=*/1);
  mingru_scan<<<4096, 96, 0, stream>>>(cbuf, c_gruw, ch, /*s_t=*/1,   /*s_f=*/128);

  // 4) grouped positional conv added into GRU outputs
  const int peBlocks = (int)(NR / 256);
  pe_add<<<peBlocks, 256, 0, stream>>>(rbuf, r_pe_w, r_pe_b, rh);
  pe_add<<<peBlocks, 256, 0, stream>>>(cbuf, c_pe_w, c_pe_b, ch);

  // 5) unpatch both paths + final sum -> fp32 output
  unpatch_out<<<1024, 256, 0, stream>>>(rh, ch, wur, wuc, r_un_b, c_un_b, out);
}
